// CoarseSNN_36498632081633
// MI455X (gfx1250) — matrix operand order repaired and checked by reading, not compiled, not run on hardware
//
#include <hip/hip_runtime.h>
#include <math.h>

typedef __attribute__((ext_vector_type(16))) _Float16 v16h;
typedef __attribute__((ext_vector_type(8)))  float    v8f;

#define TT 200
#define HH 128
#define WW 128
#define S  16          // core tile
#define R1 20          // S+4 (layer-1 state region)
#define R2 18          // S+2 (layer-2 state region)
#define XS 22          // S+6 (x input region)
#define XSZ (XS*XS)    // 484
#define NP1 (R1*R1)    // 400
#define NP2 (R2*R2)    // 324
#define C1 16
#define C2 32
#define MT2 21         // ceil(324/16)

union AFrag { v16h v; unsigned u[8]; };

// gfx1250 async global->LDS copy (ASYNCcnt-tracked), SADDR form:
// dsaddr = VDST-vgpr, mem = SGPR64 + VGPR32-offset
__device__ __forceinline__ void async_x_load(unsigned ldsOff, unsigned gByteOff,
                                             const float* base) {
  asm volatile("global_load_async_to_lds_b32 %0, %1, %2"
               :: "v"(ldsOff), "v"(gByteOff), "s"(base) : "memory");
}
__device__ __forceinline__ void wait_async0() {
  asm volatile("s_wait_asynccnt 0x0" ::: "memory");
}

__global__ __launch_bounds__(256) void snn_persistent_kernel(
    const float* __restrict__ x,
    const float* __restrict__ w1, const float* __restrict__ b1,
    const float* __restrict__ w2, const float* __restrict__ b2,
    const float* __restrict__ w3, const float* __restrict__ b3,
    float* __restrict__ out)
{
  __shared__ float xs2[2][XSZ];       // double-buffered input tile
  __shared__ float v1[NP1*C1];
  __shared__ __align__(16) _Float16 s1[NP1*C1];
  __shared__ float v2[NP2*C2];
  __shared__ __align__(16) _Float16 s2[NP2*C2];
  __shared__ float v3L[S*S*3];
  __shared__ float accL[S*S*3];
  __shared__ float b2L[C2];
  __shared__ float b3L[4];
  __shared__ __align__(32) _Float16 wB2f[10*32*16]; // conv2 B fragments: f = nt*5+kk
  __shared__ __align__(32) _Float16 wB3f[9*32*16];  // conv3 B fragments: f = tap

  const int tid  = threadIdx.x;
  const int lane = tid & 31;
  const int wave = tid >> 5;
  const int bx   = blockIdx.x;
  const int b    = bx >> 6;
  const int tile = bx & 63;
  const int oy   = (tile >> 3) * S;
  const int ox   = (tile & 7) * S;

  // ---------------- init (states zeroed every launch: determinism) ----------
  for (int i = tid; i < NP1*C1; i += 256) v1[i] = 0.f;
  for (int i = tid; i < NP2*C2; i += 256) v2[i] = 0.f;
  for (int i = tid; i < S*S*3;  i += 256) { v3L[i] = 0.f; accL[i] = 0.f; }
  if (tid < C2) b2L[tid] = b2[tid];
  if (tid < 4)  b3L[tid] = (tid < 3) ? b3[tid] : 0.f;

  // conv1 weights: each thread's channel is fixed (stride 256 == 0 mod 16),
  // so taps + bias live in registers for the whole kernel.
  const int myc = tid & 15;
  float wr[9];
  #pragma unroll
  for (int tap = 0; tap < 9; ++tap) wr[tap] = w1[myc*9 + tap];
  const float brg = b1[myc];

  // Per-thread async x-stage slots (each thread owns slot tid and tid+256).
  // Border zero-pad slots are position-invariant across t: zero both buffers once.
  const int i0 = tid / XS, j0 = tid % XS;
  const int gy0 = oy + i0 - 3, gx0 = ox + j0 - 3;
  const bool ok0 = (gy0 >= 0 && gy0 < HH && gx0 >= 0 && gx0 < WW);
  const unsigned g0 = ok0 ? (unsigned)((gy0*WW + gx0)*4) : 0u;
  const unsigned l0 = (unsigned)(size_t)&xs2[0][tid];
  const int sl1 = tid + 256;
  const int i1 = sl1 / XS, j1 = sl1 % XS;
  const int gy1 = oy + i1 - 3, gx1 = ox + j1 - 3;
  const bool has1 = (sl1 < XSZ);
  const bool ok1 = has1 && (gy1 >= 0 && gy1 < HH && gx1 >= 0 && gx1 < WW);
  const unsigned g1 = ok1 ? (unsigned)((gy1*WW + gx1)*4) : 0u;
  const unsigned l1 = has1 ? (unsigned)(size_t)&xs2[0][sl1] : 0u;
  if (!ok0)          { xs2[0][tid] = 0.f; xs2[1][tid] = 0.f; }
  if (has1 && !ok1)  { xs2[0][sl1] = 0.f; xs2[1][sl1] = 0.f; }

  // Pre-swizzle conv2 weights into B fragments (32x16, K=32), this kernel's contiguous k order:
  // lane<16 holds K=0..15, lane>=16 holds K=16..31; within: k = 2*vgpr + half.
  for (int slot = tid; slot < 10*32; slot += 256) {
    int f = slot >> 5, ln = slot & 31;
    int nt = f / 5, kk = f % 5;
    int n = (ln & 15) + nt*16;
    for (int e = 0; e < 16; ++e) {
      int kb = ((ln >= 16) ? 16 : 0) + e;     // e = 2*vgpr+half = local k
      int k  = kk*32 + kb;                    // im2col: k = tap*16 + ch
      int tap = k >> 4;
      int ch  = k & 15;
      float val = (tap < 9) ? w2[(n*C1 + ch)*9 + tap] : 0.f;  // zero-pad K 144->160
      wB2f[(f*32 + ln)*16 + e] = (_Float16)val;
    }
  }
  // conv3: K-chunk kk == tap, local k == input channel (32 channels/tap)
  for (int slot = tid; slot < 9*32; slot += 256) {
    int kk = slot >> 5, ln = slot & 31;
    int n = ln & 15;
    for (int e = 0; e < 16; ++e) {
      int ch = ((ln >= 16) ? 16 : 0) + e;
      float val = (n < 3) ? w3[(n*C2 + ch)*9 + kk] : 0.f;     // N padded 3->16
      wB3f[(kk*32 + ln)*16 + e] = (_Float16)val;
    }
  }

  // Prologue: kick off async stage of t=0 into buffer 0.
  {
    const float* xb = x + ((size_t)(b*TT + 0))*HH*WW;
    if (ok0) async_x_load(l0, g0, xb);
    if (ok1) async_x_load(l1, g1, xb);
  }
  __syncthreads();

  const int laneM = lane & 15;
  const int hi8   = (lane >> 4) << 3;   // 0 or 8; A-operand lane-half K offset is 2*hi8 (0 or 16)

  // ---------------- time loop ----------------
  for (int t = 0; t < TT; ++t) {
    const int buf = t & 1;
    // data for xs2[buf] was issued last step (or prologue): drain + publish
    wait_async0();
    __syncthreads();
    // overlap next step's HBM traffic with this entire step's compute
    if (t + 1 < TT) {
      const float* xb = x + ((size_t)(b*TT + (t+1)))*HH*WW;
      const unsigned bo = (buf ^ 1) * (unsigned)(XSZ*4);
      if (ok0) async_x_load(l0 + bo, g0, xb);
      if (ok1) async_x_load(l1 + bo, g1, xb);
    }
    const float* xcur = &xs2[buf][0];

    // phase 2: conv1 + LIF1 (scalar f32, K=9 is too small for WMMA)
    for (int o = tid; o < NP1*C1; o += 256) {
      int p = o >> 4;                         // channel is myc for every o
      int ry = p / R1, rx = p - ry*R1;
      float sum = brg;
      #pragma unroll
      for (int tap = 0; tap < 9; ++tap) {
        int dy = tap/3 - 1, dx = tap - (tap/3)*3 - 1;
        sum += wr[tap] * xcur[(ry + dy + 1)*XS + (rx + dx + 1)];
      }
      float v = 0.5f*(v1[o] + sum);
      bool sp = v >= 1.0f;
      v1[o] = sp ? 0.f : v;
      s1[o] = (_Float16)(sp ? 1.f : 0.f);
    }
    __syncthreads();

    // phase 3: conv2 + LIF2 via v_wmma_f32_16x16x32_f16 (implicit GEMM)
    for (int mt = wave; mt < MT2; mt += 8) {
      int p0 = mt*16 + laneM;
      int p  = (p0 < NP2) ? p0 : (NP2 - 1);   // clamp pad rows (results discarded)
      int py = p / R2, px = p - py*R2;
      v8f acc0 = {}; v8f acc1 = {};
      #pragma unroll
      for (int kk = 0; kk < 5; ++kk) {
        AFrag A;
        #pragma unroll
        for (int vv = 0; vv < 8; ++vv) {
          int kb = 2*hi8 + vv*2;  // A-layout local k (even)
          int k  = kk*32 + kb;
          int tap = k >> 4; if (tap > 8) tap = 8;            // pad taps hit zero B
          int ch  = k & 15;                                  // even -> ch,ch+1 same tap
          int dy = tap/3 - 1, dx = tap - (tap/3)*3 - 1;
          int sy = py + dy + 1, sx = px + dx + 1;
          A.u[vv] = *(const unsigned*)&s1[((sy*R1 + sx) << 4) + ch];
        }
        v16h B0 = *(const v16h*)&wB2f[((0*5 + kk)*32 + lane)*16];
        v16h B1 = *(const v16h*)&wB2f[((1*5 + kk)*32 + lane)*16];
        acc0 = __builtin_amdgcn_wmma_f32_16x16x32_f16(false, A.v, false, B0,
                                                      (short)0, acc0, false, false);
        acc1 = __builtin_amdgcn_wmma_f32_16x16x32_f16(false, A.v, false, B1,
                                                      (short)0, acc1, false, false);
      }
      #pragma unroll
      for (int r = 0; r < 8; ++r) {          // C/D layout: M = r + 8*(lane>=16)
        int pe = mt*16 + r + ((lane < 16) ? 0 : 8);
        if (pe < NP2) {
          int c0 = laneM;
          float vv0 = 0.5f*(v2[pe*C2 + c0] + acc0[r] + b2L[c0]);
          bool sp0 = vv0 >= 1.0f;
          v2[pe*C2 + c0] = sp0 ? 0.f : vv0;
          s2[pe*C2 + c0] = (_Float16)(sp0 ? 1.f : 0.f);
          int c1 = laneM + 16;
          float vv1 = 0.5f*(v2[pe*C2 + c1] + acc1[r] + b2L[c1]);
          bool sp1 = vv1 >= 1.0f;
          v2[pe*C2 + c1] = sp1 ? 0.f : vv1;
          s2[pe*C2 + c1] = (_Float16)(sp1 ? 1.f : 0.f);
        }
      }
    }
    __syncthreads();

    // phase 4: conv3 + LIF3 via WMMA (K = 9 taps x 32 ch = 9 chunks)
    for (int mt = wave*2; mt < wave*2 + 2; ++mt) {
      int p  = mt*16 + laneM;                 // core pixels 0..255, all valid
      int py = p >> 4, px = p & 15;
      v8f acc3 = {};
      #pragma unroll
      for (int kk = 0; kk < 9; ++kk) {
        int dy = kk/3 - 1, dx = kk - (kk/3)*3 - 1;
        int base = ((py + dy + 1)*R2 + (px + dx + 1))*C2;
        AFrag A;
        #pragma unroll
        for (int vv = 0; vv < 8; ++vv) {
          int ch = 2*hi8 + vv*2;
          A.u[vv] = *(const unsigned*)&s2[base + ch];
        }
        v16h B = *(const v16h*)&wB3f[(kk*32 + lane)*16];
        acc3 = __builtin_amdgcn_wmma_f32_16x16x32_f16(false, A.v, false, B,
                                                      (short)0, acc3, false, false);
      }
      int n = laneM;
      if (n < 3) {
        #pragma unroll
        for (int r = 0; r < 8; ++r) {
          int pe  = mt*16 + r + ((lane < 16) ? 0 : 8);
          int idx = pe*3 + n;
          float v = 0.5f*(v3L[idx] + acc3[r] + b3L[n]);
          bool sp = v >= 1.0f;
          v3L[idx]  = sp ? 0.f : v;
          accL[idx] += sp ? 1.f : 0.f;
        }
      }
    }
    __syncthreads();
  }

  // ---------------- epilogue: out = sigmoid(acc / T) ----------------
  {
    int p = tid;                 // each thread owns one core pixel
    int py = p >> 4, px = p & 15;
    #pragma unroll
    for (int c = 0; c < 3; ++c) {
      float z  = accL[p*3 + c] * (1.0f / (float)TT);
      float sg = 1.0f / (1.0f + __expf(-z));
      out[(((size_t)b*3 + c)*HH + (oy + py))*WW + (ox + px)] = sg;
    }
  }
}

extern "C" void kernel_launch(void* const* d_in, const int* in_sizes, int n_in,
                              void* d_out, int out_size, void* d_ws, size_t ws_size,
                              hipStream_t stream) {
  (void)in_sizes; (void)n_in; (void)out_size; (void)d_ws; (void)ws_size;
  const float* x  = (const float*)d_in[0];
  const float* w1 = (const float*)d_in[1];
  const float* b1 = (const float*)d_in[2];
  const float* w2 = (const float*)d_in[3];
  const float* b2 = (const float*)d_in[4];
  const float* w3 = (const float*)d_in[5];
  const float* b3 = (const float*)d_in[6];
  float* out = (float*)d_out;
  // 8 batches x 64 tiles (16x16) per 128x128 image; 256 threads = 8 wave32
  snn_persistent_kernel<<<dim3(512), dim3(256), 0, stream>>>(
      x, w1, b1, w2, b2, w3, b3, out);
}
